// SCSA_71390946394425
// MI455X (gfx1250) — matrix operand order repaired and checked by reading, not compiled, not run on hardware
//
#include <hip/hip_runtime.h>
#include <math.h>

typedef __attribute__((ext_vector_type(16))) _Float16 v16h;
typedef __attribute__((ext_vector_type(8)))  float    v8f;
typedef __attribute__((ext_vector_type(4)))  float    f32x4;

#define B_   32
#define C_   512
#define H_   56
#define W_   56
#define HN_  16
#define HD_  32
#define GC_  128
#define WIN_ 7
#define HP_  8
#define WP_  8
#define L_   64      // HP_*WP_
#define EPS_ 1e-5f
#define NEG_INF_ (-3.402823466e38f)

__device__ __forceinline__ float sigmoidf_(float x) { return 1.0f / (1.0f + __expf(-x)); }

// ---------------------------------------------------------------------------
// Kernel 1: x_h[b,c,h] = mean_w x ; x_w[b,c,w] = mean_h x. One block per (b,c).
// First touch of x: default (RT) temporal hint so the plane lands in the
// 192MB L2 and is still resident for k_gate_pool / k_final.
// ---------------------------------------------------------------------------
__global__ void k_axis_means(const float* __restrict__ x,
                             float* __restrict__ xh, float* __restrict__ xw) {
    __shared__ float plane[H_ * W_];
    const int bc = blockIdx.x;
    const f32x4* xp4 = (const f32x4*)(x + (size_t)bc * H_ * W_);
    for (int i = threadIdx.x; i < (H_ * W_) / 4; i += blockDim.x)
        ((f32x4*)plane)[i] = xp4[i];
    __syncthreads();
    const int t = threadIdx.x;
    if (t < H_) {
        float rs = 0.f, cs = 0.f;
        for (int j = 0; j < W_; ++j) {
            rs += plane[t * W_ + j];
            cs += plane[j * W_ + t];
        }
        xh[(size_t)bc * H_ + t] = rs * (1.0f / W_);
        xw[(size_t)bc * W_ + t] = cs * (1.0f / H_);
    }
}

// ---------------------------------------------------------------------------
// Kernel 2: depthwise conv1d (SAME) + GroupNorm(4) + sigmoid.
// Conv branch i == GN group i (both are 128-channel slabs).
// One block per (b, axis, branch). 256 threads.
// ---------------------------------------------------------------------------
struct ConvP { const float* w[8]; const float* b[8]; };

__global__ void k_conv_gn_sig(const float* __restrict__ xh, const float* __restrict__ xw,
                              float* __restrict__ ah, float* __restrict__ aw,
                              ConvP cp,
                              const float* __restrict__ gh_g, const float* __restrict__ gh_b,
                              const float* __restrict__ gw_g, const float* __restrict__ gw_b) {
    __shared__ float feat[GC_ * H_];   // 7168 floats
    __shared__ float red[2][256];
    const int KS[4] = {3, 5, 7, 9};
    const int bi = blockIdx.x;
    const int b = bi >> 3, r = bi & 7, axis = r >> 2, br = r & 3;
    const float* seq = axis ? xw : xh;
    float* outp      = axis ? aw : ah;
    const float* wp  = cp.w[axis * 4 + br];
    const float* bp  = cp.b[axis * 4 + br];
    const float* gg  = axis ? gw_g : gh_g;
    const float* gb  = axis ? gw_b : gh_b;
    const int k = KS[br], hk = k >> 1;

    float s = 0.f, s2 = 0.f;
    for (int idx = threadIdx.x; idx < GC_ * H_; idx += blockDim.x) {
        const int cg = idx / H_, pos = idx - cg * H_;
        const int c = br * GC_ + cg;
        const float* row = seq + ((size_t)b * C_ + c) * H_;
        float acc = bp[cg];
        for (int t = 0; t < k; ++t) {
            const int p = pos + t - hk;
            if (p >= 0 && p < H_) acc += row[p] * wp[cg * k + t];
        }
        feat[idx] = acc; s += acc; s2 += acc * acc;
    }
    red[0][threadIdx.x] = s; red[1][threadIdx.x] = s2;
    __syncthreads();
    for (int o = 128; o > 0; o >>= 1) {
        if ((int)threadIdx.x < o) {
            red[0][threadIdx.x] += red[0][threadIdx.x + o];
            red[1][threadIdx.x] += red[1][threadIdx.x + o];
        }
        __syncthreads();
    }
    const float inv  = 1.0f / (GC_ * H_);
    const float mean = red[0][0] * inv;
    const float var  = red[1][0] * inv - mean * mean;
    const float rstd = rsqrtf(var + EPS_);
    for (int idx = threadIdx.x; idx < GC_ * H_; idx += blockDim.x) {
        const int cg = idx / H_, pos = idx - cg * H_;
        const int c = br * GC_ + cg;
        const float v = (feat[idx] - mean) * rstd * gg[c] + gb[c];
        outp[((size_t)b * C_ + c) * H_ + pos] = sigmoidf_(v);
    }
}

// ---------------------------------------------------------------------------
// Kernel 3: xg = x * sig_h * sig_w (in LDS) then 7x7/7 max-pool -> y (b,c,8,8).
// One block per (b,c). x read should mostly hit L2 (RT-cached by kernel 1).
// ---------------------------------------------------------------------------
__global__ void k_gate_pool(const float* __restrict__ x, const float* __restrict__ ah,
                            const float* __restrict__ aw, float* __restrict__ y) {
    __shared__ float xg[H_ * W_];
    __shared__ float hv[H_], wv[W_];
    const int bc = blockIdx.x;
    if (threadIdx.x < H_) {
        hv[threadIdx.x] = ah[(size_t)bc * H_ + threadIdx.x];
        wv[threadIdx.x] = aw[(size_t)bc * W_ + threadIdx.x];
    }
    __syncthreads();
    const f32x4* xp4 = (const f32x4*)(x + (size_t)bc * H_ * W_);
    for (int i4 = threadIdx.x; i4 < (H_ * W_) / 4; i4 += blockDim.x) {
        const int f = i4 * 4;
        const int h = f / W_, w = f - h * W_;     // W_ % 4 == 0 -> same h in vector
        const f32x4 xv = xp4[i4];
        const float hvv = hv[h];
        f32x4 g;
        g.x = xv.x * hvv * wv[w + 0];
        g.y = xv.y * hvv * wv[w + 1];
        g.z = xv.z * hvv * wv[w + 2];
        g.w = xv.w * hvv * wv[w + 3];
        ((f32x4*)xg)[i4] = g;
    }
    __syncthreads();
    if (threadIdx.x < HP_ * WP_) {
        const int ph = threadIdx.x >> 3, pw = threadIdx.x & 7;
        float m = NEG_INF_;
        for (int dh = 0; dh < WIN_; ++dh)
            for (int dw = 0; dw < WIN_; ++dw)
                m = fmaxf(m, xg[(ph * WIN_ + dh) * W_ + pw * WIN_ + dw]);
        y[(size_t)bc * L_ + threadIdx.x] = m;
    }
}

// ---------------------------------------------------------------------------
// Kernel 4: GroupNorm(groups=1) over each batch slice of y (C_*L_ elems), in place.
// One block per b. Vectorized: c = i/64 is constant within an aligned float4.
// ---------------------------------------------------------------------------
__global__ void k_gn_full(float* __restrict__ y,
                          const float* __restrict__ g, const float* __restrict__ bta) {
    __shared__ float red[2][256];
    f32x4* yp4 = (f32x4*)(y + (size_t)blockIdx.x * C_ * L_);
    float s = 0.f, s2 = 0.f;
    for (int i4 = threadIdx.x; i4 < (C_ * L_) / 4; i4 += blockDim.x) {
        const f32x4 v = yp4[i4];
        s  += v.x + v.y + v.z + v.w;
        s2 += v.x * v.x + v.y * v.y + v.z * v.z + v.w * v.w;
    }
    red[0][threadIdx.x] = s; red[1][threadIdx.x] = s2;
    __syncthreads();
    for (int o = 128; o > 0; o >>= 1) {
        if ((int)threadIdx.x < o) {
            red[0][threadIdx.x] += red[0][threadIdx.x + o];
            red[1][threadIdx.x] += red[1][threadIdx.x + o];
        }
        __syncthreads();
    }
    const float inv  = 1.0f / (C_ * L_);
    const float mean = red[0][0] * inv;
    const float var  = red[1][0] * inv - mean * mean;
    const float rstd = rsqrtf(var + EPS_);
    for (int i4 = threadIdx.x; i4 < (C_ * L_) / 4; i4 += blockDim.x) {
        const int c = (i4 * 4) / L_;              // L_ % 4 == 0
        const float sc = rstd * g[c], of = bta[c] - mean * sc;
        f32x4 v = yp4[i4];
        v.x = v.x * sc + of; v.y = v.y * sc + of;
        v.z = v.z * sc + of; v.w = v.w * sc + of;
        yp4[i4] = v;
    }
}

// ---------------------------------------------------------------------------
// Kernel 5: per-(b,head) attention via v_wmma_f32_16x16x32_f16. One wave/block.
//   attn = softmax( (q*s) k^T ) ; o = attn v ; ca = sigmoid(mean_l o)
// Fragment layouts per CDNA5 ISA 7.12.2 (wave32):
//   A 16x32 f16 : lane=(m%16, half) ; elem e -> kloc=(v<4?0:16)+half*8+2*(v&3)+p
//   B 32x16 f16 : n=lane%16 ; elem e -> kloc = (e<8?0:8) + half*8 + e
//   C/D 16x16 f32: VGPR r -> row = 16*ti + 8*half + r, col = 16*tj + lane%16
// ---------------------------------------------------------------------------
__global__ __launch_bounds__(32) void k_attn_wmma(const float* __restrict__ y,
                                                  const float* __restrict__ wq,
                                                  const float* __restrict__ wk,
                                                  const float* __restrict__ wv,
                                                  float* __restrict__ ca) {
    __shared__ float    qs[HD_ * L_];
    __shared__ float    ks_[HD_ * L_];
    __shared__ float    vs[HD_ * L_];
    __shared__ float    at[HD_ * HD_];
    __shared__ _Float16 ah16[HD_ * HD_];
    __shared__ float    os[HD_ * L_];

    const int blk = blockIdx.x;
    const int b = blk >> 4, head = blk & 15;
    const int c0 = head * HD_;
    const int ln = threadIdx.x;
    const int m_lane = ln & 15, half = ln >> 4;
    const float scale = 0.17677669529663689f;  // HD^-0.5, folded into q

    {
        const float* yp = y + ((size_t)b * C_ + c0 + ln) * L_;
        const float fq = wq[c0 + ln] * scale, fk = wk[c0 + ln], fv = wv[c0 + ln];
        for (int l = 0; l < L_; ++l) {
            const float v = yp[l];
            qs[ln * L_ + l] = v * fq;
            ks_[ln * L_ + l] = v * fk;
            vs[ln * L_ + l] = v * fv;
        }
    }
    __syncthreads();

    // ---- GEMM 1: attn = q (32x64) . k^T  -> 2x2 tiles, K = 2 chunks of 32
    v8f acc[2][2] = {};
    for (int kk = 0; kk < 2; ++kk) {
        v16h A[2], Bm[2];
#pragma unroll
        for (int ti = 0; ti < 2; ++ti)
#pragma unroll
            for (int e = 0; e < 16; ++e) {
                const int v = e >> 1, p = e & 1;
                const int kloc = ((v < 4) ? 0 : 16) + half * 8 + 2 * (v & 3) + p;
                A[ti][e] = (_Float16)qs[(ti * 16 + m_lane) * L_ + kk * 32 + kloc];
            }
#pragma unroll
        for (int tj = 0; tj < 2; ++tj)
#pragma unroll
            for (int e = 0; e < 16; ++e)
                Bm[tj][e] = (_Float16)ks_[(tj * 16 + m_lane) * L_ + kk * 32 + ((e < 8) ? 0 : 8) + half * 8 + e];
#pragma unroll
        for (int ti = 0; ti < 2; ++ti)
#pragma unroll
            for (int tj = 0; tj < 2; ++tj)
                acc[ti][tj] = __builtin_amdgcn_wmma_f32_16x16x32_f16(
                    false, A[ti], false, Bm[tj], (short)0, acc[ti][tj], false, false);
    }
#pragma unroll
    for (int ti = 0; ti < 2; ++ti)
#pragma unroll
        for (int tj = 0; tj < 2; ++tj)
#pragma unroll
            for (int r = 0; r < 8; ++r)
                at[(ti * 16 + half * 8 + r) * HD_ + tj * 16 + m_lane] = acc[ti][tj][r];
    __syncthreads();

    // ---- softmax: lane ln owns row ln (safe in-place)
    {
        float mx = NEG_INF_;
        for (int j = 0; j < HD_; ++j) mx = fmaxf(mx, at[ln * HD_ + j]);
        float sum = 0.f;
        for (int j = 0; j < HD_; ++j) {
            const float e = __expf(at[ln * HD_ + j] - mx);
            at[ln * HD_ + j] = e; sum += e;
        }
        const float rinv = 1.0f / sum;
        for (int j = 0; j < HD_; ++j)
            ah16[ln * HD_ + j] = (_Float16)(at[ln * HD_ + j] * rinv);
    }
    __syncthreads();

    // ---- GEMM 2: o (32x64) = attn (32x32) . v (32x64), 2x4 tiles, K=32
    v8f oacc[2][4] = {};
    v16h A2[2];
#pragma unroll
    for (int ti = 0; ti < 2; ++ti)
#pragma unroll
        for (int e = 0; e < 16; ++e) {
            const int v = e >> 1, p = e & 1;
            const int kloc = ((v < 4) ? 0 : 16) + half * 8 + 2 * (v & 3) + p;
            A2[ti][e] = ah16[(ti * 16 + m_lane) * HD_ + kloc];
        }
#pragma unroll
    for (int tj = 0; tj < 4; ++tj) {
        v16h Bv;
#pragma unroll
        for (int e = 0; e < 16; ++e)
            Bv[e] = (_Float16)vs[(((e < 8) ? 0 : 8) + half * 8 + e) * L_ + tj * 16 + m_lane];
#pragma unroll
        for (int ti = 0; ti < 2; ++ti)
            oacc[ti][tj] = __builtin_amdgcn_wmma_f32_16x16x32_f16(
                false, A2[ti], false, Bv, (short)0, oacc[ti][tj], false, false);
    }
#pragma unroll
    for (int ti = 0; ti < 2; ++ti)
#pragma unroll
        for (int tj = 0; tj < 4; ++tj)
#pragma unroll
            for (int r = 0; r < 8; ++r)
                os[(ti * 16 + half * 8 + r) * L_ + tj * 16 + m_lane] = oacc[ti][tj][r];
    __syncthreads();

    {
        float s = 0.f;
        for (int l = 0; l < L_; ++l) s += os[ln * L_ + l];
        ca[(size_t)b * C_ + c0 + ln] = sigmoidf_(s * (1.0f / L_));
    }
}

// ---------------------------------------------------------------------------
// Kernel 6: out = ca[b,c] * x * sig_h[b,c,h] * sig_w[b,c,w]  (float4 path).
// Last use of x -> non-temporal load; out is a pure stream -> non-temporal
// store so the 205MB output doesn't evict L2-resident data.
// ---------------------------------------------------------------------------
__global__ void k_final(const float* __restrict__ x, const float* __restrict__ ah,
                        const float* __restrict__ aw, const float* __restrict__ ca,
                        float* __restrict__ out) {
    const size_t n4 = (size_t)B_ * C_ * H_ * W_ / 4;
    const size_t i4 = (size_t)blockIdx.x * blockDim.x + threadIdx.x;
    if (i4 >= n4) return;
    const size_t f = i4 * 4;
    const int bc = (int)(f / (H_ * W_));
    const int q = (int)(f - (size_t)bc * (H_ * W_));
    const int h = q / W_, w = q - h * W_;          // w % 4 == 0 since W_ % 4 == 0
    const f32x4 xv  = __builtin_nontemporal_load((const f32x4*)x + i4);
    const f32x4 wv4 = *(const f32x4*)(aw + (size_t)bc * W_ + w);
    const float sc  = ca[bc] * ah[(size_t)bc * H_ + h];
    f32x4 o;
    o.x = xv.x * sc * wv4.x;
    o.y = xv.y * sc * wv4.y;
    o.z = xv.z * sc * wv4.z;
    o.w = xv.w * sc * wv4.w;
    __builtin_nontemporal_store(o, (f32x4*)out + i4);
}

// ---------------------------------------------------------------------------
extern "C" void kernel_launch(void* const* d_in, const int* in_sizes, int n_in,
                              void* d_out, int out_size, void* d_ws, size_t ws_size,
                              hipStream_t stream) {
    const float* x = (const float*)d_in[0];
    ConvP cp;
    for (int i = 0; i < 8; ++i) {
        cp.w[i] = (const float*)d_in[1 + 2 * i];
        cp.b[i] = (const float*)d_in[2 + 2 * i];
    }
    const float* gn_h_g = (const float*)d_in[17];
    const float* gn_h_b = (const float*)d_in[18];
    const float* gn_w_g = (const float*)d_in[19];
    const float* gn_w_b = (const float*)d_in[20];
    const float* gn_g   = (const float*)d_in[21];
    const float* gn_b   = (const float*)d_in[22];
    const float* wq     = (const float*)d_in[23];
    const float* wk     = (const float*)d_in[24];
    const float* wv     = (const float*)d_in[25];
    float* out = (float*)d_out;

    // workspace layout (floats)
    float* ws = (float*)d_ws;
    const size_t AX = (size_t)B_ * C_ * H_;   // 917504
    float* xh = ws;                  // axis means
    float* xw = xh + AX;
    float* ah = xw + AX;             // sigmoid(GN(conv)) attns
    float* aw = ah + AX;
    float* y  = aw + AX;             // pooled (B,C,8,8)
    float* ca = y + (size_t)B_ * C_ * L_;   // (B,C)

    k_axis_means<<<B_ * C_, 256, 0, stream>>>(x, xh, xw);
    k_conv_gn_sig<<<B_ * 8, 256, 0, stream>>>(xh, xw, ah, aw, cp,
                                              gn_h_g, gn_h_b, gn_w_g, gn_w_b);
    k_gate_pool<<<B_ * C_, 256, 0, stream>>>(x, ah, aw, y);
    k_gn_full<<<B_, 256, 0, stream>>>(y, gn_g, gn_b);
    k_attn_wmma<<<B_ * HN_, 32, 0, stream>>>(y, wq, wk, wv, ca);
    const int n4 = B_ * C_ * H_ * W_ / 4;
    k_final<<<(n4 + 255) / 256, 256, 0, stream>>>(x, ah, aw, ca, out);
}
